// AFM_71588514890531
// MI455X (gfx1250) — matrix operand order repaired and checked by reading, not compiled, not run on hardware
//
#include <hip/hip_runtime.h>
#include <math.h>

// AFM forward for MI455X (gfx1250, wave32).
// Pair-attention GEMM per batch row runs on v_wmma_f32_16x16x32_bf16 in
// TRANSPOSED form:  D(16x16) = W^T(16x32) @ hb^T(32x16), pairs on the N axis.
// Each pair's 16 attention values stay within a lane's D registers (8/lane,
// other 8 in the cross half-wave lane): logit reduction = 8 in-lane FMAs +
// one v_permlanex16 (VALU crossover, no LDS). Two tiles are processed per
// loop iteration so the two WMMAs' result-consumption hazards overlap.
//
// WMMA fragment layout assumptions (CDNA5 ISA 7.12.2, wave32):
//  A (16x32 bf16): lane L: M=L&15; g=L>>4; VGPR v holds K=16g+2v, 16g+2v+1 (code's order, not ISA table).
//  B (32x16 bf16): lane L: N=L&15; g=L>>4; VGPR v holds K=16g+2v, 16g+2v+1 (code's order, not ISA table).
//  C/D (16x16 f32): VGPR i: lanes 0-15 -> M=i, lanes 16-31 -> M=8+i; N=L&15.

#define T_TREES 100
#define NPAIR   4950          // 100*99/2
#define DIM     32
#define ATT     16
#define THREADS 256
#define NWAVE   8
#define NTILE   ((NPAIR + 15) / 16)   // 310

typedef __attribute__((ext_vector_type(16))) __bf16 v16bf;
typedef __attribute__((ext_vector_type(8)))  float  v8f;

union BF16x16 { unsigned u[8]; v16bf v; };

__device__ __forceinline__ unsigned pk_bf16(float lo, float hi) {
  unsigned a = __float_as_uint(lo);
  unsigned b = __float_as_uint(hi);
  a += 0x7fffu + ((a >> 16) & 1u);   // round-to-nearest-even
  b += 0x7fffu + ((b >> 16) & 1u);
  // take bytes {b[3],b[2],a[3],a[2]} in one v_perm_b32
  return __builtin_amdgcn_perm(b, a, 0x07060302u);
}

// t + (t from lane^16), pure VALU via v_permlanex16_b32 (identity selectors)
__device__ __forceinline__ float xor16_add(float t) {
  const int o = __builtin_amdgcn_permlanex16(
      __float_as_int(t), __float_as_int(t), 0x76543210, 0xfedcba98, false, false);
  return t + __int_as_float(o);
}

__device__ __forceinline__ float waveRedMax(float v) {
  v = fmaxf(v, __shfl_xor(v, 1));
  v = fmaxf(v, __shfl_xor(v, 2));
  v = fmaxf(v, __shfl_xor(v, 4));
  v = fmaxf(v, __shfl_xor(v, 8));
  v = fmaxf(v, __shfl_xor(v, 16));
  return v;
}
__device__ __forceinline__ float waveRedSum(float v) {
  v += __shfl_xor(v, 1); v += __shfl_xor(v, 2); v += __shfl_xor(v, 4);
  v += __shfl_xor(v, 8); v += __shfl_xor(v, 16);
  return v;
}

__global__ __launch_bounds__(THREADS)
void afm_fwd_kernel(const int* __restrict__ x,
                    const float* __restrict__ Emb,
                    const float* __restrict__ W,
                    const float* __restrict__ bvec,
                    const float* __restrict__ hvec,
                    const float* __restrict__ W1,
                    const float* __restrict__ b1,
                    const float* __restrict__ W2,
                    const float* __restrict__ b2,
                    const float* __restrict__ Wf,
                    const float* __restrict__ bfin,
                    float* __restrict__ out) {
  __shared__ __align__(16) float eLds[T_TREES * DIM];     // 12.8 KB
  __shared__ __align__(16) float logitsLds[NTILE * 16];   // 19.84 KB (logits -> weights)
  __shared__ unsigned short pairRC[NPAIR];                // 9.9 KB
  __shared__ float pooledLds[DIM];
  __shared__ float redMax[NWAVE], redSum[NWAVE];
  __shared__ float o1Lds[ATT], o2Lds[8];

  const int tid  = threadIdx.x;
  const int blk  = blockIdx.x;
  const int lane = tid & 31;
  const int wave = tid >> 5;
  const int m    = lane & 15;   // A-matrix M index (att dim) / B-matrix N index (pair)
  const int g    = lane >> 4;   // half-wave group

  // ---- stage embeddings for this batch row (100 rows x 32 f32, float4 loads)
  for (int q = tid; q < T_TREES * 8; q += THREADS) {
    const int row = q >> 3, seg = q & 7;
    const long ix = (long)x[blk * T_TREES + row];
    const float4 v = *(const float4*)(Emb + ix * DIM + seg * 4);
    *(float4*)(eLds + row * DIM + seg * 4) = v;
  }
  // ---- precompute pair -> (row,col) table (triu_indices order)
  for (int p = tid; p < NPAIR; p += THREADS) {
    int r = (int)((199.0f - sqrtf(39601.0f - 8.0f * (float)p)) * 0.5f);
    r = max(0, min(r, 98));
    while (r < 98 && (99 * (r + 1) - ((r + 1) * r) / 2) <= p) ++r;
    while (r > 0  && (99 * r - (r * (r - 1)) / 2) > p) --r;
    const int base = 99 * r - (r * (r - 1)) / 2;
    const int c = p - base + r + 1;
    pairRC[p] = (unsigned short)((r << 8) | c);
  }
  if (tid < DIM) pooledLds[tid] = 0.0f;

  // ---- per-lane constants:
  // A-fragment = W^T (M=att dim m, K=embedding dim), built once.
  BF16x16 aw;
#pragma unroll
  for (int v = 0; v < 4; ++v) {
    const int k0 = 16 * g + 2 * v;
    aw.u[v] = pk_bf16(W[k0 * ATT + m], W[(k0 + 1) * ATT + m]);
  }
#pragma unroll
  for (int v = 0; v < 4; ++v) {
    const int k0 = 16 * g + 8 + 2 * v;
    aw.u[4 + v] = pk_bf16(W[k0 * ATT + m], W[(k0 + 1) * ATT + m]);
  }
  // bias/h for the 8 att dims this lane's D registers hold (M = i + 8g)
  float bh[8], hh[8];
#pragma unroll
  for (int i = 0; i < 8; ++i) {
    bh[i] = bvec[i + 8 * g];
    hh[i] = hvec[i + 8 * g];
  }

  __syncthreads();

  // ---- pass 1: attention logits via WMMA, two 16-pair tiles per iteration
  for (int tile = 2 * wave; tile < NTILE; tile += 2 * NWAVE) {
    const int pA = tile * 16 + m;             // tileB = tile+1 <= 309 always
    const int pB = pA + 16;
    const int pcA = (pA < NPAIR) ? pA : 0;    // clamp: garbage cols masked below
    const int pcB = (pB < NPAIR) ? pB : 0;
    const unsigned rcA = pairRC[pcA];
    const unsigned rcB = pairRC[pcB];
    // B-fragment needs hb[p][K] for K = 16g .. 16g+15 : 4 contiguous float4s
    const float4* erA = (const float4*)(eLds + (rcA >> 8) * DIM) + 4 * g;
    const float4* ecA = (const float4*)(eLds + (rcA & 255) * DIM) + 4 * g;
    const float4* erB = (const float4*)(eLds + (rcB >> 8) * DIM) + 4 * g;
    const float4* ecB = (const float4*)(eLds + (rcB & 255) * DIM) + 4 * g;
    const float4 a0 = erA[0], a1 = erA[1], a2 = erA[2], a3 = erA[3];
    const float4 b0 = ecA[0], b1 = ecA[1], b2 = ecA[2], b3 = ecA[3];
    const float4 c0 = erB[0], c1 = erB[1], c2 = erB[2], c3 = erB[3];
    const float4 d0 = ecB[0], d1 = ecB[1], d2 = ecB[2], d3 = ecB[3];

    BF16x16 hbA, hbB;
    hbA.u[0] = pk_bf16(a0.x * b0.x, a0.y * b0.y);
    hbA.u[1] = pk_bf16(a0.z * b0.z, a0.w * b0.w);
    hbA.u[2] = pk_bf16(a1.x * b1.x, a1.y * b1.y);
    hbA.u[3] = pk_bf16(a1.z * b1.z, a1.w * b1.w);
    hbA.u[4] = pk_bf16(a2.x * b2.x, a2.y * b2.y);
    hbA.u[5] = pk_bf16(a2.z * b2.z, a2.w * b2.w);
    hbA.u[6] = pk_bf16(a3.x * b3.x, a3.y * b3.y);
    hbA.u[7] = pk_bf16(a3.z * b3.z, a3.w * b3.w);
    hbB.u[0] = pk_bf16(c0.x * d0.x, c0.y * d0.y);
    hbB.u[1] = pk_bf16(c0.z * d0.z, c0.w * d0.w);
    hbB.u[2] = pk_bf16(c1.x * d1.x, c1.y * d1.y);
    hbB.u[3] = pk_bf16(c1.z * d1.z, c1.w * d1.w);
    hbB.u[4] = pk_bf16(c2.x * d2.x, c2.y * d2.y);
    hbB.u[5] = pk_bf16(c2.z * d2.z, c2.w * d2.w);
    hbB.u[6] = pk_bf16(c3.x * d3.x, c3.y * d3.y);
    hbB.u[7] = pk_bf16(c3.z * d3.z, c3.w * d3.w);

    v8f cz = {};
    const v8f dA = __builtin_amdgcn_wmma_f32_16x16x32_bf16(
        false, aw.v, false, hbA.v, (short)0, cz, false, false);
    const v8f dB = __builtin_amdgcn_wmma_f32_16x16x32_bf16(
        false, aw.v, false, hbB.v, (short)0, cz, false, false);

    // dX[i] = s[pair][att = i + 8g] ; logit partials fully in-lane
    float tA = 0.0f, tB = 0.0f;
#pragma unroll
    for (int i = 0; i < 8; ++i) {
      tA = fmaf(fmaxf(dA[i] + bh[i], 0.0f), hh[i], tA);
      tB = fmaf(fmaxf(dB[i] + bh[i], 0.0f), hh[i], tB);
    }
    tA = xor16_add(tA);                       // combine att dims 0-7 with 8-15
    tB = xor16_add(tB);
    if (g == 0) {
      logitsLds[pA] = (pA < NPAIR) ? tA : -INFINITY;
      logitsLds[pB] = (pB < NPAIR) ? tB : -INFINITY;
    }
  }
  __syncthreads();

  // ---- softmax over 4950 pairs (block-wide)
  float lm = -INFINITY;
  for (int p = tid; p < NPAIR; p += THREADS) lm = fmaxf(lm, logitsLds[p]);
  lm = waveRedMax(lm);
  if (lane == 0) redMax[wave] = lm;
  __syncthreads();
  float bm = redMax[0];
#pragma unroll
  for (int i = 1; i < NWAVE; ++i) bm = fmaxf(bm, redMax[i]);

  float ls = 0.0f;
  for (int p = tid; p < NPAIR; p += THREADS) {
    const float w = __expf(logitsLds[p] - bm);
    logitsLds[p] = w;                 // store unnormalized weight in place
    ls += w;
  }
  ls = waveRedSum(ls);
  if (lane == 0) redSum[wave] = ls;
  __syncthreads();
  float tot = 0.0f;
#pragma unroll
  for (int i = 0; i < NWAVE; ++i) tot += redSum[i];

  // ---- pass 2: pooled[k] = sum_p w_p * e[r][k]*e[c][k]
  {
    const int p0 = (tid * NPAIR) / THREADS;
    const int p1 = ((tid + 1) * NPAIR) / THREADS;   // contiguous chunk: reuses e[r]
    float acc[DIM];
#pragma unroll
    for (int k = 0; k < DIM; ++k) acc[k] = 0.0f;
    float4 erc[8];
    int curR = -1;
    for (int p = p0; p < p1; ++p) {
      const unsigned rc = pairRC[p];
      const int r = rc >> 8, c = rc & 255;
      const float w = logitsLds[p];
      if (r != curR) {
        const float4* er = (const float4*)(eLds + r * DIM);
#pragma unroll
        for (int q = 0; q < 8; ++q) erc[q] = er[q];
        curR = r;
      }
      const float4* ec = (const float4*)(eLds + c * DIM);
#pragma unroll
      for (int q = 0; q < 8; ++q) {
        const float4 ev = ec[q];
        acc[4 * q + 0] = fmaf(w * erc[q].x, ev.x, acc[4 * q + 0]);
        acc[4 * q + 1] = fmaf(w * erc[q].y, ev.y, acc[4 * q + 1]);
        acc[4 * q + 2] = fmaf(w * erc[q].z, ev.z, acc[4 * q + 2]);
        acc[4 * q + 3] = fmaf(w * erc[q].w, ev.w, acc[4 * q + 3]);
      }
    }
#pragma unroll
    for (int k = 0; k < DIM; ++k) atomicAdd(&pooledLds[k], acc[k]);
  }
  __syncthreads();

  // ---- normalize by softmax sum, then tiny MLP head
  if (tid < DIM) pooledLds[tid] *= (1.0f / tot);
  __syncthreads();
  if (tid < ATT) {
    float s = b1[tid];
#pragma unroll
    for (int k = 0; k < DIM; ++k) s = fmaf(pooledLds[k], W1[k * ATT + tid], s);
    o1Lds[tid] = fmaxf(s, 0.0f);
  }
  __syncthreads();
  if (tid < 8) {
    float s = b2[tid];
#pragma unroll
    for (int k = 0; k < ATT; ++k) s = fmaf(o1Lds[k], W2[k * 8 + tid], s);
    o2Lds[tid] = fmaxf(s, 0.0f);
  }
  __syncthreads();
  if (tid == 0) {
    float s = bfin[0];
#pragma unroll
    for (int k = 0; k < 8; ++k) s = fmaf(o2Lds[k], Wf[k], s);
    out[blk] = 1.0f / (1.0f + __expf(-s));
  }
}

extern "C" void kernel_launch(void* const* d_in, const int* in_sizes, int n_in,
                              void* d_out, int out_size, void* d_ws, size_t ws_size,
                              hipStream_t stream) {
  (void)in_sizes; (void)n_in; (void)d_ws; (void)ws_size;
  const int*   x    = (const int*)d_in[0];
  const float* Emb  = (const float*)d_in[1];
  const float* W    = (const float*)d_in[2];
  const float* bv   = (const float*)d_in[3];
  const float* hv   = (const float*)d_in[4];
  const float* W1   = (const float*)d_in[5];
  const float* b1   = (const float*)d_in[6];
  const float* W2   = (const float*)d_in[7];
  const float* b2   = (const float*)d_in[8];
  const float* Wf   = (const float*)d_in[9];
  const float* bfin = (const float*)d_in[10];
  float* out = (float*)d_out;
  const int bs = out_size;  // 1024 batch rows, one workgroup each
  afm_fwd_kernel<<<dim3(bs), dim3(THREADS), 0, stream>>>(
      x, Emb, W, bv, hv, W1, b1, W2, b2, Wf, bfin, out);
}
